// SuperPoint_50027779064169
// MI455X (gfx1250) — matrix operand order repaired and checked by reading, not compiled, not run on hardware
//
#include <hip/hip_runtime.h>
#include <hip/hip_bf16.h>

typedef _Float16 half_t;
typedef __attribute__((ext_vector_type(16))) _Float16 v16h;
typedef __attribute__((ext_vector_type(8)))  _Float16 v8h;
typedef __attribute__((ext_vector_type(8)))  float    v8f;
typedef __attribute__((ext_vector_type(4)))  unsigned int v4u;
typedef __attribute__((ext_vector_type(8)))  int v8i;
typedef __attribute__((ext_vector_type(4)))  int v4i;

#if defined(__HIP_DEVICE_COMPILE__) && __has_builtin(__builtin_amdgcn_tensor_load_to_lds)
#define HAVE_TDM 1
#endif

__device__ __forceinline__ void tdm_wait0()
{
#if __has_builtin(__builtin_amdgcn_s_wait_tensorcnt)
    __builtin_amdgcn_s_wait_tensorcnt((short)0);
#else
    asm volatile("s_wait_tensorcnt 0x0" ::: "memory");
#endif
}

// Issue one 2D TDM tile load: tile_w (2-byte elems) x tile_h rows, row stride
// row_stride elems, global -> LDS (contiguous). D# built per CDNA5 ISA 8.3/8.4.
__device__ __forceinline__ void tdm_load_2d(const half_t* gsrc, unsigned lds_off,
                                            int tile_w, int tile_h, int row_stride)
{
#ifdef HAVE_TDM
    unsigned long long ga = (unsigned long long)(uintptr_t)gsrc;
    v4u g0;
    g0[0] = 1u;                                           // count=1 (valid user D#)
    g0[1] = lds_off;                                      // lds_addr (bytes)
    g0[2] = (unsigned)(ga & 0xFFFFFFFFu);                 // global_addr[31:0]
    g0[3] = (unsigned)((ga >> 32) & 0x1FFFFFFu) | (2u << 30);  // addr[56:32] | type=2

    unsigned td0 = (unsigned)row_stride;                  // tensor_dim0
    unsigned td1 = (unsigned)tile_h;                      // tensor_dim1
    unsigned long long st0 = (unsigned long long)row_stride;       // dim0 stride
    unsigned long long st1 = st0 * (unsigned long long)tile_h;     // dim1 stride (2D: unused)
    v8i g1;
    g1[0] = (int)(1u << 16);                              // mask=0 | data_size=2B
    g1[1] = (int)((td0 & 0xFFFFu) << 16);                 // atomic_addr=0 | dim0[15:0]
    g1[2] = (int)(((td0 >> 16) & 0xFFFFu) | ((td1 & 0xFFFFu) << 16));
    g1[3] = (int)(((td1 >> 16) & 0xFFFFu) | (((unsigned)tile_w & 0xFFFFu) << 16)); // tile_dim0
    g1[4] = (int)((unsigned)tile_h & 0xFFFFu);            // tile_dim1 | tile_dim2=0
    g1[5] = (int)(st0 & 0xFFFFFFFFu);
    g1[6] = (int)(((st0 >> 32) & 0xFFFFu) | ((st1 & 0xFFFFu) << 16));
    g1[7] = (int)((st1 >> 16) & 0xFFFFFFFFu);

    v4i z4; z4[0] = 0; z4[1] = 0; z4[2] = 0; z4[3] = 0;   // 2D: groups 2/3 unused
#if defined(__clang_major__) && __clang_major__ >= 23
    v8i z8; 
    #pragma unroll
    for (int i = 0; i < 8; ++i) z8[i] = 0;
    __builtin_amdgcn_tensor_load_to_lds(g0, g1, z4, z4, z8, 0);
#else
    __builtin_amdgcn_tensor_load_to_lds(g0, g1, z4, z4, 0);
#endif
#else
    (void)gsrc; (void)lds_off; (void)tile_w; (void)tile_h; (void)row_stride;
#endif
}

// ---------------------------------------------------------------------------
// Implicit-GEMM conv + fused bias/ReLU/BatchNorm using v_wmma_f32_16x16x32_f16
// M = Cout (Mpad multiple of 128), N = 64 pixels of one row (4 WMMA sub-tiles),
// K = Cin*ks*ks (Kpad multiple of 32). Block = 256 threads = 8 waves.
// Weights: TDM DMAs the 128x32-half chunk for the whole block into LDS
// (double-buffered, TENSORcnt). Activations: im2col chunk gathered into LDS in
// exact B-fragment lane order (double-buffered). 4 WMMAs per wave per K-step
// reuse one A fragment; TDM + gather for chunk k+1 overlap chunk-k WMMAs.
// ---------------------------------------------------------------------------
__global__ void __launch_bounds__(256)
conv_wmma_kernel(const half_t* __restrict__ act,
                 const half_t* __restrict__ wgt,
                 const float* __restrict__ bias,
                 const float* __restrict__ gamma,
                 const float* __restrict__ beta,
                 const float* __restrict__ mean,
                 const float* __restrict__ var,
                 half_t* __restrict__ out16,
                 float* __restrict__ out32,
                 int B, int Cin, int H, int W, int Cout,
                 int ks, int Kpad, int doRelu)
{
    __shared__ __align__(32) half_t lb[2][2048];        // B: 32K x 64N halves, frag order
#ifdef HAVE_TDM
    __shared__ __align__(32) half_t wlds[2][128 * 32];  // A: 128M x 32K halves, row-major
#endif
    const int tid  = threadIdx.x;
    const int lane = tid & 31;
    const int wave = tid >> 5;

    const int tilesPerRow = (W + 63) >> 6;
    const int tile   = blockIdx.x;
    const int x0     = (tile % tilesPerRow) << 6;
    const int rowIdx = tile / tilesPerRow;
    const int y      = rowIdx % H;
    const int b      = rowIdx / H;
    const int m0     = blockIdx.y * 128 + wave * 16;

    const int Ktot = Cin * ks * ks;
    const int ks2  = ks * ks;
    const int hk   = ks >> 1;

    v8f acc0, acc1, acc2, acc3;
    #pragma unroll
    for (int i = 0; i < 8; ++i) { acc0[i] = 0.f; acc1[i] = 0.f; acc2[i] = 0.f; acc3[i] = 0.f; }

    const half_t* wblk = wgt + (size_t)blockIdx.y * 128 * Kpad;   // block's 128 weight rows
#ifndef HAVE_TDM
    const half_t* wrow = wgt + (size_t)(m0 + (lane & 15)) * Kpad + ((lane >> 4) << 3);
#endif

    // stage one 32K x 64N im2col chunk into LDS buffer `bufsel` (fragment order)
    auto stage = [&](int k0, int bufsel) {
        #pragma unroll
        for (int e0 = 0; e0 < 2048; e0 += 256) {
            int e  = e0 + tid;
            int s  = e >> 9;                 // sub-tile 0..3
            int l  = (e >> 4) & 31;          // target lane
            int t  = e & 15;                 // half slot within lane
            int Kr = (t & 7) + ((l >> 4) << 3) + ((t >> 3) << 4);    // K = (t&7) + 8*(lane>=16) + 16*(t>=8)
            int n  = (s << 4) + (l & 15);    // N within the 64-pixel tile
            int kk = k0 + Kr;
            half_t v = (half_t)0.0f;
            if (kk < Ktot) {
                int cin = kk / ks2;
                int r   = kk - cin * ks2;
                int ry  = r / ks;
                int yy  = y + ry - hk;
                int xx  = x0 + n + (r - ry * ks) - hk;
                if (yy >= 0 && yy < H && xx >= 0 && xx < W)
                    v = act[(((size_t)b * Cin + cin) * H + yy) * W + xx];
            }
            lb[bufsel][e] = v;
        }
    };

    // ---- prologue: stage chunk 0 (B via gather, A via TDM) ----
    stage(0, 0);
#ifdef HAVE_TDM
    if (wave == 0) {
        tdm_load_2d(wblk, (unsigned)(uintptr_t)&wlds[0][0], 32, 128, Kpad);
        tdm_wait0();
    }
#endif
    __syncthreads();
    int cur = 0;

    for (int k0 = 0; k0 < Ktot; k0 += 32) {
        const int nxt = k0 + 32;
        const bool have = nxt < Ktot;
        if (have) {
            stage(nxt, cur ^ 1);                          // overlap gather with WMMAs
#ifdef HAVE_TDM
            if (wave == 0)
                tdm_load_2d(wblk + nxt, (unsigned)(uintptr_t)&wlds[cur ^ 1][0],
                            32, 128, Kpad);               // overlap DMA with WMMAs
#endif
        }

        // ---- A fragment (per ISA 16-bit A layout: M = lane&15,
        //      lanes 0-15 -> K {0..7,16..23}, lanes 16-31 -> K {8..15,24..31}) ----
#ifdef HAVE_TDM
        const half_t* wp = &wlds[cur][((wave << 4) + (lane & 15)) * 32 + ((lane >> 4) << 3)];
#else
        const half_t* wp = wrow + k0;
        __builtin_prefetch(wp + 64, 0, 1);                // global_prefetch_b8
#endif
        v8h alo = *reinterpret_cast<const v8h*>(wp);      // halves 0..7
        v8h ahi = *reinterpret_cast<const v8h*>(wp + 16); // halves 8..15
        v16h a;
        #pragma unroll
        for (int i = 0; i < 8; ++i) { a[i] = alo[i]; a[i + 8] = ahi[i]; }

        // ---- 4 B fragments: 32B contiguous LDS reads per lane ----
        const half_t* lbc = &lb[cur][lane << 4];
        v16h b0 = *reinterpret_cast<const v16h*>(lbc + 0);
        v16h b1 = *reinterpret_cast<const v16h*>(lbc + 512);
        v16h b2 = *reinterpret_cast<const v16h*>(lbc + 1024);
        v16h b3 = *reinterpret_cast<const v16h*>(lbc + 1536);

        acc0 = __builtin_amdgcn_wmma_f32_16x16x32_f16(false, a, false, b0, (short)0, acc0, false, false);
        acc1 = __builtin_amdgcn_wmma_f32_16x16x32_f16(false, a, false, b1, (short)0, acc1, false, false);
        acc2 = __builtin_amdgcn_wmma_f32_16x16x32_f16(false, a, false, b2, (short)0, acc2, false, false);
        acc3 = __builtin_amdgcn_wmma_f32_16x16x32_f16(false, a, false, b3, (short)0, acc3, false, false);

#ifdef HAVE_TDM
        if (wave == 0 && have) tdm_wait0();               // next weight chunk landed
#endif
        __syncthreads();                                  // buffers swap-safe
        cur ^= 1;
    }

    // ---- epilogue: bias + ReLU + BN(eval), C/D layout: M = r + 8*(lane>=16) ----
    const int nl = lane & 15;
    #pragma unroll
    for (int r = 0; r < 8; ++r) {
        int m = m0 + r + ((lane >> 4) << 3);
        if (m < Cout) {
            float bb = bias[m];
            float sc = gamma[m] * rsqrtf(var[m] + 1e-3f);
            float mm = mean[m];
            float be = beta[m];
            float vs[4] = { acc0[r], acc1[r], acc2[r], acc3[r] };
            #pragma unroll
            for (int s = 0; s < 4; ++s) {
                int x = x0 + (s << 4) + nl;
                if (x < W) {
                    float v = vs[s] + bb;
                    if (doRelu) v = fmaxf(v, 0.0f);
                    v = (v - mm) * sc + be;
                    size_t o = (((size_t)b * Cout + m) * H + y) * W + x;
                    if (out16) out16[o] = (half_t)v;
                    if (out32) out32[o] = v;
                }
            }
        }
    }
}

// ---- pack fp32 OIHW weights into f16 [Mpad x Kpad] GEMM matrix (zero pad) ----
__global__ void prep_weights_kernel(const float* __restrict__ w, half_t* __restrict__ o,
                                    int Cout, int Cin, int ks, int Mpad, int Kpad)
{
    int i = blockIdx.x * blockDim.x + threadIdx.x;
    int tot = Mpad * Kpad;
    if (i >= tot) return;
    int m = i / Kpad, k = i - m * Kpad;
    int Kt = Cin * ks * ks;
    half_t v = (half_t)0.0f;
    if (m < Cout && k < Kt) v = (half_t)w[(size_t)m * Kt + k];  // OIHW flat == im2col K order
    o[(size_t)m * Kpad + k] = v;
}

__global__ void cvt_f32_to_f16(const float* __restrict__ in, half_t* __restrict__ out, int n)
{
    int i = blockIdx.x * blockDim.x + threadIdx.x;
    if (i < n) out[i] = (half_t)in[i];
}

__global__ void maxpool2_kernel(const half_t* __restrict__ in, half_t* __restrict__ out,
                                int B, int C, int Hi, int Wi)
{
    int Ho = Hi >> 1, Wo = Wi >> 1;
    int i = blockIdx.x * blockDim.x + threadIdx.x;
    int tot = B * C * Ho * Wo;
    if (i >= tot) return;
    int xo = i % Wo, rest = i / Wo;
    int yo = rest % Ho, bc = rest / Ho;
    const half_t* p = in + (((size_t)bc * Hi + yo * 2) * Wi + xo * 2);
    float m = fmaxf(fmaxf((float)p[0], (float)p[1]),
                    fmaxf((float)p[Wi], (float)p[Wi + 1]));
    out[i] = (half_t)m;
}

// softmax over 65 cells, drop dustbin, depth-to-space 8x8 -> scores [B,640,640]
__global__ void softmax_d2s_kernel(const float* __restrict__ det, float* __restrict__ S,
                                   int B, int h, int w)
{
    int p = blockIdx.x * blockDim.x + threadIdx.x;
    int tot = B * h * w;
    if (p >= tot) return;
    int b = p / (h * w), pix = p % (h * w);
    int y = pix / w, x = pix % w;
    const float* q = det + (size_t)b * 65 * h * w + pix;
    float mx = -3.0e38f;
    for (int c = 0; c < 65; ++c) mx = fmaxf(mx, q[(size_t)c * h * w]);
    float sum = 0.0f;
    for (int c = 0; c < 65; ++c) sum += __expf(q[(size_t)c * h * w] - mx);
    float inv = 1.0f / sum;
    int Wf = w * 8;
    for (int c = 0; c < 64; ++c) {
        float v = __expf(q[(size_t)c * h * w] - mx) * inv;
        int yy = y * 8 + (c >> 3);
        int xx = x * 8 + (c & 7);
        S[(size_t)b * (h * 8) * Wf + (size_t)yy * Wf + xx] = v;
    }
}

// separable 9-tap max (radius 4, SAME with -inf identity == clamp-to-valid)
__global__ void hmax9_kernel(const float* __restrict__ in, float* __restrict__ out,
                             int B, int H, int W)
{
    int i = blockIdx.x * blockDim.x + threadIdx.x;
    int tot = B * H * W;
    if (i >= tot) return;
    int x = i % W;
    const float* row = in + (i - x);
    int lo = x - 4 < 0 ? 0 : x - 4;
    int hi = x + 4 >= W ? W - 1 : x + 4;
    float m = -3.0e38f;
    for (int xx = lo; xx <= hi; ++xx) m = fmaxf(m, row[xx]);
    out[i] = m;
}

__global__ void vmax9_kernel(const float* __restrict__ in, float* __restrict__ out,
                             int B, int H, int W)
{
    int i = blockIdx.x * blockDim.x + threadIdx.x;
    int tot = B * H * W;
    if (i >= tot) return;
    int x = i % W, rest = i / W;
    int y = rest % H, b = rest / H;
    int lo = y - 4 < 0 ? 0 : y - 4;
    int hi = y + 4 >= H ? H - 1 : y + 4;
    const float* col = in + (size_t)b * H * W + x;
    float m = -3.0e38f;
    for (int yy = lo; yy <= hi; ++yy) m = fmaxf(m, col[(size_t)yy * W]);
    out[i] = m;
}

__global__ void ew_eq_kernel(const float* S, const float* P, float* M, int n)
{
    int i = blockIdx.x * blockDim.x + threadIdx.x;
    if (i < n) M[i] = (S[i] == P[i]) ? 1.0f : 0.0f;
}
__global__ void ew_supp_kernel(const float* P, const float* S, float* SUPP, float* SS, int n)
{
    int i = blockIdx.x * blockDim.x + threadIdx.x;
    if (i >= n) return;
    float sp = (P[i] > 0.0f) ? 1.0f : 0.0f;
    SUPP[i] = sp;
    SS[i] = (sp > 0.0f) ? 0.0f : S[i];
}
__global__ void ew_upd_kernel(const float* SS, const float* P, const float* SUPP, float* M, int n)
{
    int i = blockIdx.x * blockDim.x + threadIdx.x;
    if (i >= n) return;
    if (SUPP[i] == 0.0f && SS[i] == P[i]) M[i] = 1.0f;
}
__global__ void ew_final_kernel(const float* M, const float* S, float* SC, int B, int H, int W)
{
    int i = blockIdx.x * blockDim.x + threadIdx.x;
    int tot = B * H * W;
    if (i >= tot) return;
    int x = i % W, rest = i / W, y = rest % H;
    float v = (M[i] > 0.0f) ? S[i] : 0.0f;
    if (y < 4 || x < 4 || y >= H - 4 || x >= W - 4) v = -1.0f;
    SC[i] = v;
}

// per-batch iterative top-K (destroys sc, which is recomputed scratch each call)
__global__ void __launch_bounds__(1024)
topk_kernel(float* __restrict__ sc, float* __restrict__ kpts, float* __restrict__ tops,
            float* __restrict__ valid, int HW, int Wimg, int Kk)
{
    int b = blockIdx.x;
    float* s = sc + (size_t)b * HW;
    __shared__ float sv[1024];
    __shared__ int   si[1024];
    int tid = threadIdx.x;
    for (int k = 0; k < Kk; ++k) {
        float best = -3.0e38f; int bi = 0;
        for (int i = tid; i < HW; i += 1024) {
            float v = s[i];
            if (v > best) { best = v; bi = i; }
        }
        sv[tid] = best; si[tid] = bi;
        __syncthreads();
        for (int off = 512; off > 0; off >>= 1) {
            if (tid < off && sv[tid + off] > sv[tid]) { sv[tid] = sv[tid + off]; si[tid] = si[tid + off]; }
            __syncthreads();
        }
        if (tid == 0) {
            float v = sv[0]; int idx = si[0];
            tops[b * Kk + k] = v;
            kpts[(size_t)(b * Kk + k) * 2 + 0] = (float)(idx % Wimg);
            kpts[(size_t)(b * Kk + k) * 2 + 1] = (float)(idx / Wimg);
            valid[b * Kk + k] = (v > 0.005f) ? 1.0f : 0.0f;
            s[idx] = -3.0e38f;
        }
        __syncthreads();
    }
}

// per-pixel L2 normalize over C=256 channels
__global__ void __launch_bounds__(256)
l2norm_kernel(const float* __restrict__ in, float* __restrict__ out, int C, int hw)
{
    int p = blockIdx.x;
    int b = p / hw, pix = p - b * hw;
    int c = threadIdx.x;
    size_t idx = ((size_t)b * C + c) * hw + pix;
    float v = in[idx];
    __shared__ float red[256];
    red[c] = v * v;
    __syncthreads();
    for (int o = 128; o > 0; o >>= 1) {
        if (c < o) red[c] += red[c + o];
        __syncthreads();
    }
    float nrm = fmaxf(sqrtf(red[0]), 1e-12f);
    out[idx] = v / nrm;
}

// bilinear grid_sample (align_corners=True semantics of the reference) + renorm
__global__ void __launch_bounds__(256)
sample_desc_kernel(const float* __restrict__ dn, const float* __restrict__ kpts,
                   float* __restrict__ desc, int C, int h, int w, int Kk)
{
    int bk = blockIdx.x;
    int b = bk / Kk;
    float kx = kpts[(size_t)bk * 2 + 0];
    float ky = kpts[(size_t)bk * 2 + 1];
    const float s = 8.0f;
    float dxn = w * s - s * 0.5f - 0.5f;
    float dyn = h * s - s * 0.5f - 0.5f;
    float gx = ((kx - s * 0.5f + 0.5f) / dxn) * 2.0f - 1.0f;
    float gy = ((ky - s * 0.5f + 0.5f) / dyn) * 2.0f - 1.0f;
    float ix = (gx + 1.0f) * 0.5f * (w - 1);
    float iy = (gy + 1.0f) * 0.5f * (h - 1);
    float x0f = floorf(ix), y0f = floorf(iy);
    float wx = ix - x0f, wy = iy - y0f;
    int x0 = min(max((int)x0f, 0), w - 1);
    int x1 = min(max((int)x0f + 1, 0), w - 1);
    int y0 = min(max((int)y0f, 0), h - 1);
    int y1 = min(max((int)y0f + 1, 0), h - 1);
    int c = threadIdx.x;
    const float* base = dn + ((size_t)b * C + c) * h * w;
    float v = base[(size_t)y0 * w + x0] * (1 - wx) * (1 - wy)
            + base[(size_t)y0 * w + x1] * wx * (1 - wy)
            + base[(size_t)y1 * w + x0] * (1 - wx) * wy
            + base[(size_t)y1 * w + x1] * wx * wy;
    __shared__ float red[256];
    red[c] = v * v;
    __syncthreads();
    for (int o = 128; o > 0; o >>= 1) {
        if (c < o) red[c] += red[c + o];
        __syncthreads();
    }
    float nrm = fmaxf(sqrtf(red[0]), 1e-12f);
    desc[(size_t)bk * C + c] = v / nrm;
}

// ---------------------------------------------------------------------------
extern "C" void kernel_launch(void* const* d_in, const int* in_sizes, int n_in,
                              void* d_out, int out_size, void* d_ws, size_t ws_size,
                              hipStream_t stream)
{
    (void)in_sizes; (void)n_in; (void)out_size; (void)ws_size;
    const float* image = (const float*)d_in[0];

    struct CSpec { int cin, cout, ks; };
    static const CSpec CS[12] = {
        {1,64,3},{64,64,3},{64,64,3},{64,64,3},
        {64,128,3},{128,128,3},{128,128,3},{128,128,3},
        {128,256,3},{256,65,1},{128,256,3},{256,256,1}};

    char* base = (char*)d_ws;
    size_t off = 0;
    auto alloc = [&](size_t bytes) { size_t o = off; off = (off + bytes + 255) & ~(size_t)255; return o; };

    const int B = 2;
    const size_t actElems = (size_t)B * 64 * 640 * 640;
    half_t* actA = (half_t*)(base + alloc(actElems * 2));
    half_t* actB = (half_t*)(base + alloc(actElems * 2));

    half_t* wptr[12]; int Mpad[12], Kpad[12];
    for (int i = 0; i < 12; ++i) {
        int Kt = CS[i].cin * CS[i].ks * CS[i].ks;
        Kpad[i] = ((Kt + 31) / 32) * 32;
        Mpad[i] = ((CS[i].cout + 127) / 128) * 128;
        wptr[i] = (half_t*)(base + alloc((size_t)Mpad[i] * Kpad[i] * 2));
    }
    float* dmap  = (float*)(base + alloc((size_t)B * 256 * 80 * 80 * 4));
    float* dnorm = (float*)(base + alloc((size_t)B * 256 * 80 * 80 * 4));
    float* det2  = (float*)(base + alloc((size_t)B * 65 * 80 * 80 * 4));
    const size_t simg = (size_t)B * 640 * 640;
    float* S    = (float*)(base + alloc(simg * 4));
    float* T    = (float*)(base + alloc(simg * 4));
    float* P    = (float*)(base + alloc(simg * 4));
    float* M    = (float*)(base + alloc(simg * 4));
    float* SUPP = (float*)(base + alloc(simg * 4));
    float* SS   = (float*)(base + alloc(simg * 4));
    float* SC   = (float*)(base + alloc(simg * 4));

    float* out     = (float*)d_out;
    float* o_kpts  = out;                       // [2,1024,2]
    float* o_desc  = out + 4096;                // [2,1024,256]
    float* o_tops  = out + 4096 + 524288;       // [2,1024]
    float* o_valid = o_tops + 2048;             // [2,1024]

    // pack weights, convert image
    for (int i = 0; i < 12; ++i) {
        const float* w = (const float*)d_in[1 + 6 * i + 0];
        int n = Mpad[i] * Kpad[i];
        prep_weights_kernel<<<(n + 255) / 256, 256, 0, stream>>>(
            w, wptr[i], CS[i].cout, CS[i].cin, CS[i].ks, Mpad[i], Kpad[i]);
    }
    {
        int n = B * 640 * 640;
        cvt_f32_to_f16<<<(n + 255) / 256, 256, 0, stream>>>(image, actA, n);
    }

    auto conv = [&](int li, const half_t* in, half_t* o16, float* o32, int Hc, int Wc) {
        const CSpec& c = CS[li];
        int tilesPerRow = (Wc + 63) / 64;
        dim3 grid((unsigned)(B * Hc * tilesPerRow), (unsigned)(Mpad[li] / 128));
        int relu = (li == 9 || li == 11) ? 0 : 1;
        conv_wmma_kernel<<<grid, 256, 0, stream>>>(
            in, wptr[li],
            (const float*)d_in[1 + 6 * li + 1], (const float*)d_in[1 + 6 * li + 2],
            (const float*)d_in[1 + 6 * li + 3], (const float*)d_in[1 + 6 * li + 4],
            (const float*)d_in[1 + 6 * li + 5],
            o16, o32, B, c.cin, Hc, Wc, c.cout, c.ks, Kpad[li], relu);
    };
    auto pool2 = [&](const half_t* in, half_t* o, int C, int Hi, int Wi) {
        int n = B * C * (Hi / 2) * (Wi / 2);
        maxpool2_kernel<<<(n + 255) / 256, 256, 0, stream>>>(in, o, B, C, Hi, Wi);
    };

    conv(0, actA, actB, nullptr, 640, 640);
    conv(1, actB, actA, nullptr, 640, 640);
    pool2(actA, actB, 64, 640, 640);
    conv(2, actB, actA, nullptr, 320, 320);
    conv(3, actA, actB, nullptr, 320, 320);
    pool2(actB, actA, 64, 320, 320);
    conv(4, actA, actB, nullptr, 160, 160);
    conv(5, actB, actA, nullptr, 160, 160);
    pool2(actA, actB, 128, 160, 160);
    conv(6, actB, actA, nullptr, 80, 80);
    conv(7, actA, actB, nullptr, 80, 80);          // x in actB
    conv(10, actB, actA, nullptr, 80, 80);         // descriptor conv3x3
    conv(11, actA, nullptr, dmap, 80, 80);         // descriptor 1x1 -> f32
    conv(8, actB, actA, nullptr, 80, 80);          // detector conv3x3
    conv(9, actA, nullptr, det2, 80, 80);          // detector 1x1 -> f32

    {
        int n = B * 80 * 80;
        softmax_d2s_kernel<<<(n + 255) / 256, 256, 0, stream>>>(det2, S, B, 80, 80);
    }

    const int nel = (int)simg;
    auto pool9 = [&](const float* in, float* outp) {
        hmax9_kernel<<<(nel + 255) / 256, 256, 0, stream>>>(in, T, B, 640, 640);
        vmax9_kernel<<<(nel + 255) / 256, 256, 0, stream>>>(T, outp, B, 640, 640);
    };
    pool9(S, P);
    ew_eq_kernel<<<(nel + 255) / 256, 256, 0, stream>>>(S, P, M, nel);
    for (int it = 0; it < 2; ++it) {
        pool9(M, P);
        ew_supp_kernel<<<(nel + 255) / 256, 256, 0, stream>>>(P, S, SUPP, SS, nel);
        pool9(SS, P);
        ew_upd_kernel<<<(nel + 255) / 256, 256, 0, stream>>>(SS, P, SUPP, M, nel);
    }
    ew_final_kernel<<<(nel + 255) / 256, 256, 0, stream>>>(M, S, SC, B, 640, 640);

    topk_kernel<<<B, 1024, 0, stream>>>(SC, o_kpts, o_tops, o_valid, 640 * 640, 640, 1024);

    l2norm_kernel<<<B * 80 * 80, 256, 0, stream>>>(dmap, dnorm, 256, 80 * 80);
    sample_desc_kernel<<<B * 1024, 256, 0, stream>>>(dnorm, o_kpts, o_desc, 256, 80, 80, 1024);
}
